// SparseRelationalEncoder_28209345200433
// MI455X (gfx1250) — matrix operand order repaired and checked by reading, not compiled, not run on hardware
//
#include <hip/hip_runtime.h>
#include <math.h>

typedef __bf16 bf16_t;
typedef __attribute__((ext_vector_type(16))) __bf16 v16bf;
typedef __attribute__((ext_vector_type(8)))  float  v8f;

#define NNODES 128
#define NEDGES 16256
#define BATCH  8
#define BE_ROWS (BATCH * NEDGES)   // 130048
#define BN_ROWS (BATCH * NNODES)   // 1024
#define HID 256
#define STAT_THREADS 65536

__device__ __forceinline__ float elu_f(float x) {
  return x > 0.f ? x : (__expf(x) - 1.f);
}

// ---------------------------------------------------------------------------
// Weight convert: W f32 [Ksrc x Nsrc] row-major -> Wt bf16 [Npad x Kpad]
// (transposed, zero padded). B-fragments then read contiguous K-pairs.
// ---------------------------------------------------------------------------
__global__ void __launch_bounds__(256)
conv_wt(const float* __restrict__ W, bf16_t* __restrict__ Wt,
        int Ksrc, int Kpad, int Nsrc, int Npad) {
  int i = blockIdx.x * 256 + threadIdx.x;
  if (i >= Npad * Kpad) return;
  int n = i / Kpad;
  int k = i - n * Kpad;
  float v = (k < Ksrc && n < Nsrc) ? W[(size_t)k * Nsrc + n] : 0.f;
  Wt[i] = (bf16_t)v;
}

// inputs f32 [1024 x 100] -> bf16 [1024 x 128] zero padded
__global__ void __launch_bounds__(256)
pad_input(const float* __restrict__ x, bf16_t* __restrict__ A) {
  int i = blockIdx.x * 256 + threadIdx.x;
  if (i >= BN_ROWS * 128) return;
  int r = i >> 7;
  int k = i & 127;
  A[i] = (bf16_t)(k < 100 ? x[r * 100 + k] : 0.f);
}

// ---------------------------------------------------------------------------
// WMMA GEMM: out = ELU(A[MxK] @ Wt^T + bias), bf16 in, f32 accumulate.
// MODE 0: plain A (lda given)
// MODE 1: A row m=(b,e) is concat( xnode[b,recv(e)], xnode[b,send(e)] ), K=512
// MODE 2: concat( xnode[b,recv], xnode[b,send], xskip[b,e] ), K=768
// Block: 256 thr = 8 waves; block tile 128(M) x 128(N); wave tile 32 x 64
// (2 A-frags x 4 B-frags -> 8 v_wmma per K-step of 32).
// M must be a multiple of 128; K a multiple of 32; Wt has >= gridDim.y*128 rows.
// ---------------------------------------------------------------------------
template <int MODE>
__global__ void __launch_bounds__(256)
gemm_bias_elu(const bf16_t* __restrict__ A, int lda,
              const bf16_t* __restrict__ xnode, const bf16_t* __restrict__ xskip,
              const bf16_t* __restrict__ Wt, int K,
              const float* __restrict__ bias, int Nout, int ldo,
              bf16_t* __restrict__ out_bf, float* __restrict__ out_f) {
  const int lane = threadIdx.x & 31;
  const int wave = threadIdx.x >> 5;
  const int wm = wave & 3, wn = wave >> 2;
  const int mbase = blockIdx.x * 128 + wm * 32;
  const int nbase = blockIdx.y * 128 + wn * 64;
  const int hi  = lane >> 4;
  const int l15 = lane & 15;

  // Per-lane A row base pointers (one per 256-wide K segment).
  const bf16_t* as0[2];
  const bf16_t* as1[2];
  const bf16_t* as2[2];
#pragma unroll
  for (int mi = 0; mi < 2; ++mi) {
    int m = mbase + mi * 16 + l15;
    if (MODE == 0) {
      as0[mi] = A + (size_t)m * lda;
      as1[mi] = as0[mi];
      as2[mi] = as0[mi];
    } else {
      int b    = m / NEDGES;
      int e    = m - b * NEDGES;
      int recv = e / (NNODES - 1);
      int rr   = e - recv * (NNODES - 1);
      int send = rr + (rr >= recv ? 1 : 0);
      as0[mi] = xnode + ((size_t)(b * NNODES + recv)) * HID;
      as1[mi] = xnode + ((size_t)(b * NNODES + send)) * HID;
      as2[mi] = (MODE == 2) ? (xskip + (size_t)m * HID) : as1[mi];
    }
  }
  const bf16_t* brow0 = Wt + (size_t)(nbase +  0 + l15) * K;
  const bf16_t* brow1 = Wt + (size_t)(nbase + 16 + l15) * K;
  const bf16_t* brow2 = Wt + (size_t)(nbase + 32 + l15) * K;
  const bf16_t* brow3 = Wt + (size_t)(nbase + 48 + l15) * K;

  v8f acc00 = {}, acc01 = {}, acc02 = {}, acc03 = {};
  v8f acc10 = {}, acc11 = {}, acc12 = {}, acc13 = {};

  for (int k0 = 0; k0 < K; k0 += 32) {
    const int seg = (MODE == 0) ? 0 : (k0 >> 8);
    const int kk  = (MODE == 0) ? k0 : (k0 & 255);
    const bf16_t* pa0 =
        ((MODE == 0) ? as0[0]
                     : (seg == 0 ? as0[0] : (seg == 1 ? as1[0] : as2[0]))) + kk;
    const bf16_t* pa1 =
        ((MODE == 0) ? as0[1]
                     : (seg == 0 ? as0[1] : (seg == 1 ? as1[1] : as2[1]))) + kk;

    v16bf a0, a1, b0, b1, b2, b3;
    // A fragment (16-bit 16x32 layout): lane half picks K+8; VGPRs 4-7 add K+16
#pragma unroll
    for (int v = 0; v < 8; ++v) {
      int k = ((v & 4) << 2) + hi * 8 + 2 * (v & 3);
      a0[2 * v]     = pa0[k];
      a0[2 * v + 1] = pa0[k + 1];
      a1[2 * v]     = pa1[k];
      a1[2 * v + 1] = pa1[k + 1];
    }
    // B fragment (16-bit 32x16): same K order as A (lane half picks K+8; VGPRs 4-7 add K+16)
    const bf16_t* pb0 = brow0 + k0;
    const bf16_t* pb1 = brow1 + k0;
    const bf16_t* pb2 = brow2 + k0;
    const bf16_t* pb3 = brow3 + k0;
#pragma unroll
    for (int v = 0; v < 8; ++v) {
      int k = ((v & 4) << 2) + hi * 8 + 2 * (v & 3);
      b0[2 * v] = pb0[k]; b0[2 * v + 1] = pb0[k + 1];
      b1[2 * v] = pb1[k]; b1[2 * v + 1] = pb1[k + 1];
      b2[2 * v] = pb2[k]; b2[2 * v + 1] = pb2[k + 1];
      b3[2 * v] = pb3[k]; b3[2 * v + 1] = pb3[k + 1];
    }
    acc00 = __builtin_amdgcn_wmma_f32_16x16x32_bf16(false, a0, false, b0, (short)0, acc00, false, false);
    acc01 = __builtin_amdgcn_wmma_f32_16x16x32_bf16(false, a0, false, b1, (short)0, acc01, false, false);
    acc02 = __builtin_amdgcn_wmma_f32_16x16x32_bf16(false, a0, false, b2, (short)0, acc02, false, false);
    acc03 = __builtin_amdgcn_wmma_f32_16x16x32_bf16(false, a0, false, b3, (short)0, acc03, false, false);
    acc10 = __builtin_amdgcn_wmma_f32_16x16x32_bf16(false, a1, false, b0, (short)0, acc10, false, false);
    acc11 = __builtin_amdgcn_wmma_f32_16x16x32_bf16(false, a1, false, b1, (short)0, acc11, false, false);
    acc12 = __builtin_amdgcn_wmma_f32_16x16x32_bf16(false, a1, false, b2, (short)0, acc12, false, false);
    acc13 = __builtin_amdgcn_wmma_f32_16x16x32_bf16(false, a1, false, b3, (short)0, acc13, false, false);
  }

  auto epi = [&](const v8f& acc, int mi, int ni) {
    int col = nbase + ni * 16 + l15;
    if (col >= Nout) return;
    float bv = bias[col];
    float r[8];
#pragma unroll
    for (int v = 0; v < 8; ++v) r[v] = elu_f(acc[v] + bv);
    size_t rowb = (size_t)(mbase + mi * 16 + 8 * hi);
    if (out_bf) {
#pragma unroll
      for (int v = 0; v < 8; ++v) out_bf[(rowb + v) * ldo + col] = (bf16_t)r[v];
    }
    if (out_f) {
#pragma unroll
      for (int v = 0; v < 8; ++v) out_f[(rowb + v) * ldo + col] = r[v];
    }
  };
  epi(acc00, 0, 0); epi(acc01, 0, 1); epi(acc02, 0, 2); epi(acc03, 0, 3);
  epi(acc10, 1, 0); epi(acc11, 1, 1); epi(acc12, 1, 2); epi(acc13, 1, 3);
}

// ---------------------------------------------------------------------------
// BatchNorm stats, deterministic two-stage reduction over M rows x C channels.
// ---------------------------------------------------------------------------
__global__ void __launch_bounds__(256)
bn_stats1(const float* __restrict__ h, int M, int C, int ld,
          float* __restrict__ psum, float* __restrict__ psq) {
  int g = blockIdx.x * 256 + threadIdx.x;   // g in [0, STAT_THREADS)
  int c = g % C;
  int r0 = g / C;
  int rstride = STAT_THREADS / C;
  float s = 0.f, q = 0.f;
  for (int r = r0; r < M; r += rstride) {
    float v = h[(size_t)r * ld + c];
    s += v;
    q += v * v;
  }
  psum[g] = s;
  psq[g]  = q;
}

__global__ void __launch_bounds__(256)
bn_stats2(const float* __restrict__ psum, const float* __restrict__ psq,
          int C, float invM,
          const float* __restrict__ gamma, const float* __restrict__ beta,
          float* __restrict__ scale, float* __restrict__ shift) {
  int c = blockIdx.x;
  __shared__ float ss[256];
  __shared__ float sq[256];
  float s = 0.f, q = 0.f;
  int nper = STAT_THREADS / C;
  for (int i = threadIdx.x; i < nper; i += 256) {
    int g = i * C + c;
    s += psum[g];
    q += psq[g];
  }
  ss[threadIdx.x] = s;
  sq[threadIdx.x] = q;
  __syncthreads();
  for (int o = 128; o > 0; o >>= 1) {
    if (threadIdx.x < o) {
      ss[threadIdx.x] += ss[threadIdx.x + o];
      sq[threadIdx.x] += sq[threadIdx.x + o];
    }
    __syncthreads();
  }
  if (threadIdx.x == 0) {
    float mean = ss[0] * invM;
    float var  = sq[0] * invM - mean * mean;
    float sc   = gamma[c] * rsqrtf(var + 1e-5f);
    scale[c] = sc;
    shift[c] = beta[c] - mean * sc;
  }
}

__global__ void __launch_bounds__(256)
bn_apply(const float* __restrict__ h, int n, int C,
         const float* __restrict__ scale, const float* __restrict__ shift,
         bf16_t* __restrict__ out_bf, float* __restrict__ out_f, int relu) {
  int i = blockIdx.x * 256 + threadIdx.x;
  if (i >= n) return;
  int c = i % C;
  float v = h[i] * scale[c] + shift[c];
  if (relu) v = fmaxf(v, 0.f);
  if (out_bf) out_bf[i] = (bf16_t)v;
  if (out_f)  out_f[i]  = v;
}

// ---------------------------------------------------------------------------
// edge2node: out[b,n,c] = (1/128) * sum_{r<127} x[b, n*127+r, c]
// (edges with recv==n are the contiguous block [n*127, (n+1)*127))
// ---------------------------------------------------------------------------
__global__ void __launch_bounds__(256)
edge2node(const bf16_t* __restrict__ xe, bf16_t* __restrict__ out) {
  int i = blockIdx.x * 256 + threadIdx.x;  // i = (b*128+n)*256 + c
  if (i >= BN_ROWS * HID) return;
  int c   = i & 255;
  int row = i >> 8;
  int b   = row >> 7;
  int n   = row & 127;
  const bf16_t* p = xe + ((size_t)(b * NEDGES + n * (NNODES - 1))) * HID + c;
  float s = 0.f;
  for (int r = 0; r < NNODES - 1; ++r) s += (float)p[(size_t)r * HID];
  out[i] = (bf16_t)(s * (1.0f / NNODES));
}

// ---------------------------------------------------------------------------
extern "C" void kernel_launch(void* const* d_in, const int* in_sizes, int n_in,
                              void* d_out, int out_size, void* d_ws, size_t ws_size,
                              hipStream_t stream) {
  (void)in_sizes; (void)n_in; (void)out_size; (void)ws_size;
  const float* inp = (const float*)d_in[0];
  // d_in[1]=rel_rec, d_in[2]=rel_send: one-hot over ordered off-diagonal pairs;
  // indices are computed analytically in-kernel, so these are unused.
  const float *W1[7], *B1[7], *W2[7], *B2[7], *GA[7], *BT[7];
  for (int i = 0; i < 7; ++i) {
    W1[i] = (const float*)d_in[3 + i * 6 + 0];
    B1[i] = (const float*)d_in[3 + i * 6 + 1];
    W2[i] = (const float*)d_in[3 + i * 6 + 2];
    B2[i] = (const float*)d_in[3 + i * 6 + 3];
    GA[i] = (const float*)d_in[3 + i * 6 + 4];
    BT[i] = (const float*)d_in[3 + i * 6 + 5];
  }

  // ---- workspace carve (256B aligned) ----
  char* base = (char*)d_ws;
  size_t off = 0;
  auto alloc = [&](size_t bytes) -> void* {
    off = (off + 255) & ~(size_t)255;
    void* p = base + off;
    off += bytes;
    return p;
  };
  // transposed bf16 weights: [Npad x Kpad]
  struct WD { int ks, kp, ns, np; };
  const WD wd1[7] = {{100,128,256,256},{512,512,256,256},{256,256,256,256},
                     {768,768,256,256},{256,256,256,256},{256,256,256,256},
                     {256,256,256,256}};
  const WD wd2[7] = {{256,256,256,256},{256,256,256,256},{256,256,256,256},
                     {256,256,256,256},{256,256,2,128},{256,256,2,128},{256,256,2,128}};
  bf16_t *wt1[7], *wt2[7];
  for (int i = 0; i < 7; ++i) wt1[i] = (bf16_t*)alloc((size_t)wd1[i].np * wd1[i].kp * 2);
  for (int i = 0; i < 7; ++i) wt2[i] = (bf16_t*)alloc((size_t)wd2[i].np * wd2[i].kp * 2);

  float* psum  = (float*)alloc(STAT_THREADS * 4);
  float* psq   = (float*)alloc(STAT_THREADS * 4);
  float* scale = (float*)alloc(HID * 4);
  float* shift = (float*)alloc(HID * 4);

  bf16_t* a1pad   = (bf16_t*)alloc((size_t)BN_ROWS * 128 * 2);
  bf16_t* t_small = (bf16_t*)alloc((size_t)BN_ROWS * HID * 2);
  float*  h_small = (float*)alloc((size_t)BN_ROWS * HID * 4);  // also head h [BE x 2]
  bf16_t* x1      = (bf16_t*)alloc((size_t)BN_ROWS * HID * 2);
  bf16_t* xin3    = (bf16_t*)alloc((size_t)BN_ROWS * HID * 2);
  bf16_t* x3      = (bf16_t*)alloc((size_t)BN_ROWS * HID * 2);
  bf16_t* t_big   = (bf16_t*)alloc((size_t)BE_ROWS * HID * 2);
  float*  h_big   = (float*)alloc((size_t)BE_ROWS * HID * 4);
  bf16_t* x2      = (bf16_t*)alloc((size_t)BE_ROWS * HID * 2); // x_skip; reused as x4

  // ---- weight conversion ----
  for (int i = 0; i < 7; ++i) {
    int n1 = wd1[i].np * wd1[i].kp;
    conv_wt<<<(n1 + 255) / 256, 256, 0, stream>>>(W1[i], wt1[i], wd1[i].ks, wd1[i].kp, wd1[i].ns, wd1[i].np);
    int n2 = wd2[i].np * wd2[i].kp;
    conv_wt<<<(n2 + 255) / 256, 256, 0, stream>>>(W2[i], wt2[i], wd2[i].ks, wd2[i].kp, wd2[i].ns, wd2[i].np);
  }
  pad_input<<<(BN_ROWS * 128) / 256, 256, 0, stream>>>(inp, a1pad);

  auto bn = [&](const float* h, int M, int C, int ld, int mlp) {
    bn_stats1<<<STAT_THREADS / 256, 256, 0, stream>>>(h, M, C, ld, psum, psq);
    bn_stats2<<<C, 256, 0, stream>>>(psum, psq, C, 1.0f / (float)M, GA[mlp], BT[mlp], scale, shift);
  };

  const dim3 gSmall(BN_ROWS / 128, HID / 128);  // (8, 2)
  const dim3 gBig(BE_ROWS / 128, HID / 128);    // (1016, 2)
  const dim3 gHead2(BE_ROWS / 128, 1);          // Nout padded to 128

  // ---- mlp1 : [1024 x 100] ----
  gemm_bias_elu<0><<<gSmall, 256, 0, stream>>>(a1pad, 128, nullptr, nullptr, wt1[0], 128, B1[0], HID, HID, t_small, nullptr);
  gemm_bias_elu<0><<<gSmall, 256, 0, stream>>>(t_small, HID, nullptr, nullptr, wt2[0], HID, B2[0], HID, HID, nullptr, h_small);
  bn(h_small, BN_ROWS, HID, HID, 0);
  bn_apply<<<(BN_ROWS * HID) / 256, 256, 0, stream>>>(h_small, BN_ROWS * HID, HID, scale, shift, x1, nullptr, 0);

  // ---- mlp2 : node2edge fused, K=512 ----
  gemm_bias_elu<1><<<gBig, 256, 0, stream>>>(nullptr, 0, x1, nullptr, wt1[1], 512, B1[1], HID, HID, t_big, nullptr);
  gemm_bias_elu<0><<<gBig, 256, 0, stream>>>(t_big, HID, nullptr, nullptr, wt2[1], HID, B2[1], HID, HID, nullptr, h_big);
  bn(h_big, BE_ROWS, HID, HID, 1);
  bn_apply<<<(BE_ROWS * HID) / 256, 256, 0, stream>>>(h_big, BE_ROWS * HID, HID, scale, shift, x2, nullptr, 0);

  // ---- edge2node ----
  edge2node<<<(BN_ROWS * HID) / 256, 256, 0, stream>>>(x2, xin3);

  // ---- mlp3 ----
  gemm_bias_elu<0><<<gSmall, 256, 0, stream>>>(xin3, HID, nullptr, nullptr, wt1[2], HID, B1[2], HID, HID, t_small, nullptr);
  gemm_bias_elu<0><<<gSmall, 256, 0, stream>>>(t_small, HID, nullptr, nullptr, wt2[2], HID, B2[2], HID, HID, nullptr, h_small);
  bn(h_small, BN_ROWS, HID, HID, 2);
  bn_apply<<<(BN_ROWS * HID) / 256, 256, 0, stream>>>(h_small, BN_ROWS * HID, HID, scale, shift, x3, nullptr, 0);

  // ---- mlp4 : node2edge + skip fused, K=768; ReLU after BN ----
  gemm_bias_elu<2><<<gBig, 256, 0, stream>>>(nullptr, 0, x3, x2, wt1[3], 768, B1[3], HID, HID, t_big, nullptr);
  gemm_bias_elu<0><<<gBig, 256, 0, stream>>>(t_big, HID, nullptr, nullptr, wt2[3], HID, B2[3], HID, HID, nullptr, h_big);
  bn(h_big, BE_ROWS, HID, HID, 3);
  bn_apply<<<(BE_ROWS * HID) / 256, 256, 0, stream>>>(h_big, BE_ROWS * HID, HID, scale, shift, x2, nullptr, 1); // x2 reused as x4

  // ---- output heads (Nout=2, W2 padded to 128 rows) ----
  for (int k = 0; k < 3; ++k) {
    int p = 4 + k;
    gemm_bias_elu<0><<<gBig, 256, 0, stream>>>(x2, HID, nullptr, nullptr, wt1[p], HID, B1[p], HID, HID, t_big, nullptr);
    gemm_bias_elu<0><<<gHead2, 256, 0, stream>>>(t_big, HID, nullptr, nullptr, wt2[p], HID, B2[p], 2, 2, nullptr, h_small);
    bn(h_small, BE_ROWS, 2, 2, p);
    bn_apply<<<(BE_ROWS * 2) / 256, 256, 0, stream>>>(h_small, BE_ROWS * 2, 2, scale, shift, nullptr,
                                                      (float*)d_out + (size_t)k * BE_ROWS * 2, 0);
  }
}
